// HeteroGNN_51745765982901
// MI455X (gfx1250) — matrix operand order repaired and checked by reading, not compiled, not run on hardware
//
#include <hip/hip_runtime.h>
#include <cstddef>

// ---------------------------------------------------------------------------
// Problem sizes (match reference)
// ---------------------------------------------------------------------------
#define NG 50000
#define NC 100000
#define DIN 128
#define DH 64
#define EGG 800000
#define ECG 1600000
#define ECC 1600000

typedef __attribute__((ext_vector_type(16))) __bf16 v16bf;
typedef __attribute__((ext_vector_type(8)))  float  v8f;

// ---------------------------------------------------------------------------
// WMMA GEMM: out[N,64] = act(X[N,K] @ W[K,64] (+ bias))
// bf16 inputs staged in LDS, f32 accumulate via v_wmma_f32_16x16x32_bf16.
// 64 rows per block, 128 threads (4 waves), wave w owns rows [16w,16w+16).
// Safe for in-place (out == X): a block only reads/writes its own 64 rows,
// reads happen fully into LDS before any write.
// ---------------------------------------------------------------------------
template <int K, bool BIAS, bool RELU>
__global__ __launch_bounds__(128) void gemm64_bf16wmma(
    const float* __restrict__ X, const float* __restrict__ W,
    const float* __restrict__ bias, float* __restrict__ out, int N)
{
    __shared__ __bf16 sA[64][K];   // X tile, row-major [m][k]
    __shared__ __bf16 sB[64][K];   // W transposed:  sB[n][k] = W[k][n]
    __shared__ float  sC[64][68];  // f32 result staging (padded)

    const int t  = threadIdx.x;
    const int r0 = blockIdx.x * 64;

    // Stage X tile (zero-pad rows past N) and W^T into LDS as bf16.
    for (int idx = t; idx < 64 * K; idx += 128) {
        const int r = idx / K, k = idx - r * K;
        const float v = (r0 + r < N) ? X[(size_t)(r0 + r) * K + k] : 0.0f;
        sA[r][k] = (__bf16)v;
    }
    for (int idx = t; idx < K * 64; idx += 128) {
        const int k = idx >> 6, n = idx & 63;
        sB[n][k] = (__bf16)W[idx];
    }
    __syncthreads();

    const int wave = t >> 5;
    const int lane = t & 31;
    const int hi   = lane >> 4;   // 0: lanes 0-15, 1: lanes 16-31
    const int l15  = lane & 15;
    const int mrow = wave * 16 + l15;

    v8f acc0 = {}, acc1 = {}, acc2 = {}, acc3 = {};

#pragma unroll
    for (int kt = 0; kt < K / 32; ++kt) {
        // A fragment: 16x32 bf16. Lane<16 holds K {0..7,16..23}, lane>=16
        // holds K {8..15,24..31} (ISA 7.12.2): elem j -> K = j + (j&8) + 8*hi.
        v16bf a;
        const int ka = kt * 32 + hi * 8;
#pragma unroll
        for (int j = 0; j < 16; ++j) a[j] = sA[mrow][ka + j + (j & 8)];

        // B fragment: 32x16 bf16. Lane = column N (mod 16); same K order as A:
        // elem j -> K = j + (j&8) + 8*hi.
        const int kb = kt * 32 + hi * 8;
        v16bf b0, b1, b2, b3;
#pragma unroll
        for (int j = 0; j < 16; ++j) {
            b0[j] = sB[l15][kb + j + (j & 8)];
            b1[j] = sB[16 + l15][kb + j + (j & 8)];
            b2[j] = sB[32 + l15][kb + j + (j & 8)];
            b3[j] = sB[48 + l15][kb + j + (j & 8)];
        }
        acc0 = __builtin_amdgcn_wmma_f32_16x16x32_bf16(false, a, false, b0, (short)0, acc0, false, false);
        acc1 = __builtin_amdgcn_wmma_f32_16x16x32_bf16(false, a, false, b1, (short)0, acc1, false, false);
        acc2 = __builtin_amdgcn_wmma_f32_16x16x32_bf16(false, a, false, b2, (short)0, acc2, false, false);
        acc3 = __builtin_amdgcn_wmma_f32_16x16x32_bf16(false, a, false, b3, (short)0, acc3, false, false);
    }

    // C layout: VGPR r, lane -> M = r + 8*hi, N = lane&15 (ISA 7.12.2).
#pragma unroll
    for (int r = 0; r < 8; ++r) {
        const int m = wave * 16 + hi * 8 + r;
        sC[m][l15]      = acc0[r];
        sC[m][16 + l15] = acc1[r];
        sC[m][32 + l15] = acc2[r];
        sC[m][48 + l15] = acc3[r];
    }
    __syncthreads();

    for (int idx = t; idx < 64 * 64; idx += 128) {
        const int r = idx >> 6, c = idx & 63;
        if (r0 + r < N) {
            float v = sC[r][c];
            if (BIAS) v += bias[c];
            if (RELU) v = fmaxf(v, 0.0f);
            out[(size_t)(r0 + r) * 64 + c] = v;
        }
    }
}

// ---------------------------------------------------------------------------
// LayerNorm(row of 64) + ReLU, one wave per row (wave32, lane handles 2 elems)
// ---------------------------------------------------------------------------
__global__ __launch_bounds__(128) void ln_relu_rows(
    const float* __restrict__ in, float* __restrict__ out,
    const float* __restrict__ sc, const float* __restrict__ bi, int N)
{
    const int row = blockIdx.x * 4 + (threadIdx.x >> 5);
    if (row >= N) return;
    const int lane = threadIdx.x & 31;
    const float* x = in + (size_t)row * 64;
    const float x0 = x[lane], x1 = x[lane + 32];
    float s = x0 + x1, s2 = x0 * x0 + x1 * x1;
#pragma unroll
    for (int m = 16; m >= 1; m >>= 1) {
        s  += __shfl_xor(s,  m, 32);
        s2 += __shfl_xor(s2, m, 32);
    }
    const float mean = s * (1.0f / 64.0f);
    const float var  = s2 * (1.0f / 64.0f) - mean * mean;
    const float rstd = rsqrtf(var + 1e-5f);
    float* o = out + (size_t)row * 64;
    o[lane]      = fmaxf(0.0f, (x0 - mean) * rstd * sc[lane]      + bi[lane]);
    o[lane + 32] = fmaxf(0.0f, (x1 - mean) * rstd * sc[lane + 32] + bi[lane + 32]);
}

// g = scatter + dinv^2*xt + b_gcn + gin ; then shared LN + ReLU -> out
__global__ __launch_bounds__(128) void gcn_combine_ln_relu(
    const float* __restrict__ scat, const float* __restrict__ xt,
    const float* __restrict__ dinv, const float* __restrict__ bg,
    const float* __restrict__ gin,
    const float* __restrict__ sc, const float* __restrict__ bi,
    float* __restrict__ out, int N)
{
    const int row = blockIdx.x * 4 + (threadIdx.x >> 5);
    if (row >= N) return;
    const int lane = threadIdx.x & 31;
    const size_t base = (size_t)row * 64;
    const float d2 = dinv[row] * dinv[row];
    const float x0 = scat[base + lane]      + d2 * xt[base + lane]      + bg[lane]      + gin[base + lane];
    const float x1 = scat[base + lane + 32] + d2 * xt[base + lane + 32] + bg[lane + 32] + gin[base + lane + 32];
    float s = x0 + x1, s2 = x0 * x0 + x1 * x1;
#pragma unroll
    for (int m = 16; m >= 1; m >>= 1) {
        s  += __shfl_xor(s,  m, 32);
        s2 += __shfl_xor(s2, m, 32);
    }
    const float mean = s * (1.0f / 64.0f);
    const float var  = s2 * (1.0f / 64.0f) - mean * mean;
    const float rstd = rsqrtf(var + 1e-5f);
    float* o = out + base;
    o[lane]      = fmaxf(0.0f, (x0 - mean) * rstd * sc[lane]      + bi[lane]);
    o[lane + 32] = fmaxf(0.0f, (x1 - mean) * rstd * sc[lane + 32] + bi[lane + 32]);
}

// ---------------------------------------------------------------------------
// Scatter kernels: one wave per edge, float2 per lane, f32 global atomics
// ---------------------------------------------------------------------------
__global__ __launch_bounds__(256) void gin_scatter(
    const int* __restrict__ row, const int* __restrict__ col,
    const float* __restrict__ x, float* __restrict__ out, int E)
{
    const int e = blockIdx.x * 8 + (threadIdx.x >> 5);
    if (e >= E) return;
    const int lane = threadIdx.x & 31;
    const int r = row[e], c = col[e];
    const float* src = x + (size_t)r * 64 + lane * 2;
    float* dst = out + (size_t)c * 64 + lane * 2;
    atomicAdd(dst,     src[0]);
    atomicAdd(dst + 1, src[1]);
}

__global__ __launch_bounds__(256) void gcn_edge_scatter(
    const int* __restrict__ row, const int* __restrict__ col,
    const float* __restrict__ ew, const float* __restrict__ dinv,
    const float* __restrict__ xt, float* __restrict__ out, int E)
{
    const int e = blockIdx.x * 8 + (threadIdx.x >> 5);
    if (e >= E) return;
    const int lane = threadIdx.x & 31;
    const int r = row[e], c = col[e];
    const float w = dinv[r] * ew[e] * dinv[c];
    const float* src = xt + (size_t)r * 64 + lane * 2;
    float* dst = out + (size_t)c * 64 + lane * 2;
    atomicAdd(dst,     w * src[0]);
    atomicAdd(dst + 1, w * src[1]);
}

__global__ __launch_bounds__(256) void deg_scatter(
    const int* __restrict__ col, const float* __restrict__ ew,
    float* __restrict__ deg, int E)
{
    const int e = blockIdx.x * 256 + threadIdx.x;
    if (e < E) atomicAdd(&deg[col[e]], ew[e]);
}

// deg -> dinv in place: rsqrt(deg + 1) (self-loop); deg+1 >= 1 always
__global__ __launch_bounds__(256) void dinv_kernel(float* __restrict__ d, int N)
{
    const int i = blockIdx.x * 256 + threadIdx.x;
    if (i < N) d[i] = rsqrtf(d[i] + 1.0f);
}

// h = (1+eps)*x_dst + h   (GIN pre-MLP), eps is a 1-element device array
__global__ __launch_bounds__(256) void gin_h_kernel(
    const float* __restrict__ xdst, const float* __restrict__ eps,
    float* __restrict__ h, int n)
{
    const int i = blockIdx.x * 256 + threadIdx.x;
    if (i < n) h[i] = (1.0f + eps[0]) * xdst[i] + h[i];
}

// logits = xg @ W_out[64,3] + b ; softmax over 3
__global__ __launch_bounds__(256) void head_softmax(
    const float* __restrict__ xg, const float* __restrict__ Wo,
    const float* __restrict__ bo, float* __restrict__ out, int N)
{
    const int row = blockIdx.x * 256 + threadIdx.x;
    if (row >= N) return;
    const float* x = xg + (size_t)row * 64;
    float l0 = bo[0], l1 = bo[1], l2 = bo[2];
#pragma unroll 8
    for (int d = 0; d < 64; ++d) {
        const float v = x[d];
        l0 += v * Wo[d * 3 + 0];
        l1 += v * Wo[d * 3 + 1];
        l2 += v * Wo[d * 3 + 2];
    }
    const float m = fmaxf(l0, fmaxf(l1, l2));
    const float e0 = expf(l0 - m), e1 = expf(l1 - m), e2 = expf(l2 - m);
    const float inv = 1.0f / (e0 + e1 + e2);
    out[(size_t)row * 3 + 0] = e0 * inv;
    out[(size_t)row * 3 + 1] = e1 * inv;
    out[(size_t)row * 3 + 2] = e2 * inv;
}

// ---------------------------------------------------------------------------
// Launcher
// ---------------------------------------------------------------------------
extern "C" void kernel_launch(void* const* d_in, const int* in_sizes, int n_in,
                              void* d_out, int out_size, void* d_ws, size_t ws_size,
                              hipStream_t stream)
{
    (void)in_sizes; (void)n_in; (void)out_size; (void)ws_size;

    const float* x_glom = (const float*)d_in[0];
    const float* x_cell = (const float*)d_in[1];
    const int*   gg     = (const int*)d_in[2];
    const float* ew     = (const float*)d_in[3];
    const int*   cgE    = (const int*)d_in[4];
    const int*   ccE    = (const int*)d_in[5];
    const float* W_in_g = (const float*)d_in[6];
    const float* b_in_g = (const float*)d_in[7];
    const float* lnig_s = (const float*)d_in[8];
    const float* lnig_b = (const float*)d_in[9];
    const float* W_in_c = (const float*)d_in[10];
    const float* b_in_c = (const float*)d_in[11];
    const float* lnic_s = (const float*)d_in[12];
    const float* lnic_b = (const float*)d_in[13];

    const float *W_gcn[2], *b_gcn[2], *eps_cg[2], *W_cg[2], *b_cg[2],
                *eps_cc[2], *W_cc[2], *b_cc[2], *ln_s[2], *ln_b[2];
    for (int l = 0; l < 2; ++l) {
        const int base = 14 + l * 10;
        W_gcn[l]  = (const float*)d_in[base + 0];
        b_gcn[l]  = (const float*)d_in[base + 1];
        eps_cg[l] = (const float*)d_in[base + 2];
        W_cg[l]   = (const float*)d_in[base + 3];
        b_cg[l]   = (const float*)d_in[base + 4];
        eps_cc[l] = (const float*)d_in[base + 5];
        W_cc[l]   = (const float*)d_in[base + 6];
        b_cc[l]   = (const float*)d_in[base + 7];
        ln_s[l]   = (const float*)d_in[base + 8];
        ln_b[l]   = (const float*)d_in[base + 9];
    }
    const float* W_out = (const float*)d_in[34];
    const float* b_out = (const float*)d_in[35];

    // Workspace layout (all f32): ~102.6 MB
    float* ws = (float*)d_ws;
    const size_t NG64 = (size_t)NG * 64, NC64 = (size_t)NC * 64;
    float* xg  = ws;            // [NG,64] glom features
    float* xc  = xg  + NG64;    // [NC,64] cell features
    float* tg1 = xc  + NC64;    // [NG,64] GCN xt / fc-glom pre-LN
    float* tg2 = tg1 + NG64;    // [NG,64] GCN edge-scatter accum
    float* tg3 = tg2 + NG64;    // [NG,64] GIN cg agg/h/out
    float* tc1 = tg3 + NG64;    // [NC,64] GIN cc agg/h/out / fc-cell pre-LN
    float* dv  = tc1 + NC64;    // [NG]    degree -> dinv (in place)

    const int* gg_row = gg,  *gg_col = gg  + EGG;
    const int* cg_row = cgE, *cg_col = cgE + ECG;
    const int* cc_row = ccE, *cc_col = ccE + ECC;

    const int GB_G = (NG + 63) / 64, GB_C = (NC + 63) / 64;  // gemm blocks
    const int RB_G = (NG + 3) / 4,   RB_C = (NC + 3) / 4;    // row-wave blocks

    // ---- input FC -> LN -> ReLU per node type ----
    gemm64_bf16wmma<DIN, true, false><<<GB_G, 128, 0, stream>>>(x_glom, W_in_g, b_in_g, tg1, NG);
    ln_relu_rows<<<RB_G, 128, 0, stream>>>(tg1, xg, lnig_s, lnig_b, NG);
    gemm64_bf16wmma<DIN, true, false><<<GB_C, 128, 0, stream>>>(x_cell, W_in_c, b_in_c, tc1, NC);
    ln_relu_rows<<<RB_C, 128, 0, stream>>>(tc1, xc, lnic_s, lnic_b, NC);

    // ---- two hetero message-passing layers ----
    for (int l = 0; l < 2; ++l) {
        // GCN on glom-glom: xt = xg @ W ; degree norm ; edge scatter
        gemm64_bf16wmma<DH, false, false><<<GB_G, 128, 0, stream>>>(xg, W_gcn[l], nullptr, tg1, NG);
        (void)hipMemsetAsync(dv, 0, (size_t)NG * sizeof(float), stream);
        deg_scatter<<<(EGG + 255) / 256, 256, 0, stream>>>(gg_col, ew, dv, EGG);
        dinv_kernel<<<(NG + 255) / 256, 256, 0, stream>>>(dv, NG);
        (void)hipMemsetAsync(tg2, 0, NG64 * sizeof(float), stream);
        gcn_edge_scatter<<<(EGG + 7) / 8, 256, 0, stream>>>(gg_row, gg_col, ew, dv, tg1, tg2, EGG);

        // GIN cell->glom: agg, h=(1+eps)*xg+agg, relu(h@W+b)  (in-place gemm)
        (void)hipMemsetAsync(tg3, 0, NG64 * sizeof(float), stream);
        gin_scatter<<<(ECG + 7) / 8, 256, 0, stream>>>(cg_row, cg_col, xc, tg3, ECG);
        gin_h_kernel<<<(int)((NG64 + 255) / 256), 256, 0, stream>>>(xg, eps_cg[l], tg3, (int)NG64);
        gemm64_bf16wmma<DH, true, true><<<GB_G, 128, 0, stream>>>(tg3, W_cg[l], b_cg[l], tg3, NG);

        // GIN cell->cell
        (void)hipMemsetAsync(tc1, 0, NC64 * sizeof(float), stream);
        gin_scatter<<<(ECC + 7) / 8, 256, 0, stream>>>(cc_row, cc_col, xc, tc1, ECC);
        gin_h_kernel<<<(int)((NC64 + 255) / 256), 256, 0, stream>>>(xc, eps_cc[l], tc1, (int)NC64);
        gemm64_bf16wmma<DH, true, true><<<GB_C, 128, 0, stream>>>(tc1, W_cc[l], b_cc[l], tc1, NC);

        // combine glom relations + shared LN + ReLU ; cell LN + ReLU
        // (all readers of old xg/xc have completed by stream order)
        gcn_combine_ln_relu<<<RB_G, 128, 0, stream>>>(tg2, tg1, dv, b_gcn[l], tg3,
                                                      ln_s[l], ln_b[l], xg, NG);
        ln_relu_rows<<<RB_C, 128, 0, stream>>>(tc1, xc, ln_s[l], ln_b[l], NC);
    }

    // ---- output head: logits + softmax ----
    head_softmax<<<(NG + 255) / 256, 256, 0, stream>>>(xg, W_out, b_out, (float*)d_out, NG);
}
